// LNN_87548613361825
// MI455X (gfx1250) — matrix operand order repaired and checked by reading, not compiled, not run on hardware
//
#include <hip/hip_runtime.h>

typedef __attribute__((ext_vector_type(16))) _Float16 v16h;
typedef __attribute__((ext_vector_type(8)))  float    v8f;

#define TILES 2048   // B=32768 / 16 elements per tile

__device__ __forceinline__ v8f wmma16(v16h a, v16h b, v8f c) {
  // D(f32 16x16) = A(f16 16x32) * B(f16 32x16) + C
  return __builtin_amdgcn_wmma_f32_16x16x32_f16(false, a, false, b, (short)0, c, false, false);
}

// Branchless tanh: 1 - 2/(exp2(2*log2e*x)+1)  -> v_exp_f32 + v_rcp_f32, no divergence
__device__ __forceinline__ float fast_tanh(float x) {
  float e = __builtin_amdgcn_exp2f(x * 2.88539008177793f);  // exp(2x)
  return 1.f - 2.f * __builtin_amdgcn_rcpf(e + 1.f);
}

// Address of element (K,N) inside a K=256 x N=16 B-operand panel stored as
// 8 fragments of 512 halves (fragment f: lane*16+h, lane=N+16*(Kmod32>=16), h=K&15)
__device__ __forceinline__ int baddr(int K, int N) {
  return ((K >> 4) << 8) + (N << 4) + (K & 15);
}
// Address inside one A-operand 16x32 tile for element (mm,kk)
__device__ __forceinline__ int aaddr(int mm, int kk) {
  return ((mm + ((kk >> 4) << 4)) << 4) + (kk & 15);
}

__global__ __launch_bounds__(128, 1)
void lnn_kernel(const float* __restrict__ x,  const float* __restrict__ W1,
                const float* __restrict__ b1, const float* __restrict__ W2,
                const float* __restrict__ b2, const float* __restrict__ W3,
                float* __restrict__ out)
{
  // ---- LDS: ~322.5 KB of the WGP's 320 KiB ----
  __shared__ __attribute__((aligned(32))) _Float16 sW2A[65536]; // W2  in A-layout tiles (m,f)
  __shared__ __attribute__((aligned(32))) _Float16 sW2T[65536]; // W2^T in A-layout tiles
  __shared__ __attribute__((aligned(32))) _Float16 sW1c[4096];  // W1  A-tiles, K=16 compressed
  __shared__ __attribute__((aligned(32))) _Float16 sW1T[4096];  // W1^T A-fragments (M=16, K=256)
  __shared__ __attribute__((aligned(32))) _Float16 sH1[4096];   // h1 panel (B-layout, 256x16)
  __shared__ __attribute__((aligned(32))) _Float16 sB [4096];   // scratch B-panel / f32 solve buffer
  __shared__ __attribute__((aligned(32))) _Float16 sC2[4096];   // c2[i][n] = -2*w3s*h2*s2
  __shared__ __attribute__((aligned(32))) _Float16 sCQ[4096];   // cQ[i][n] = -2*h1*s1*q
  __shared__ float sHcol[2048];                                 // per-element Hessian columns
  __shared__ float sb1[256], sb2[256], sw3[256];

  const int t    = threadIdx.x;
  const int lane = t & 31;
  const int wv   = t >> 5;
  const int n16  = lane & 15;
  const int hi8  = (lane >> 4) << 3;

  // ---------- stage weights into LDS in WMMA operand layouts ----------
  for (int idx = t; idx < 65536; idx += 128) {
    int i = idx >> 8, j = idx & 255;           // W2[i][j]
    _Float16 v = (_Float16)W2[idx];
    sW2A[((((i >> 4) << 3) + (j >> 5)) << 9) + aaddr(i & 15, j & 31)] = v;
    sW2T[((((j >> 4) << 3) + (i >> 5)) << 9) + aaddr(j & 15, i & 31)] = v;
  }
  for (int idx = t; idx < 4096; idx += 128) {
    int i = idx >> 4, k = idx & 15;            // W1[i][k]
    _Float16 v = (_Float16)W1[idx];
    sW1c[((i >> 4) << 8) + (((i & 15) + (((k >> 3) & 1) << 4)) << 3) + (k & 7)] = v;
    sW1T[((i >> 5) << 9) + aaddr(k, i & 31)] = v;
  }
  for (int i = t; i < 256; i += 128) {
    sb1[i] = b1[i]; sb2[i] = b2[i]; sw3[i] = W3[i] + W3[256 + i];
  }
  __syncthreads();

  for (int tile = blockIdx.x; tile < TILES; tile += gridDim.x) {
    // ---------- phase 0: x B-panel (K=16 dims zero-padded to 32, N=16 batch) ----------
    sB[256 + 2 * t] = (_Float16)0.f;
    sB[257 + 2 * t] = (_Float16)0.f;
    #pragma unroll
    for (int rep = 0; rep < 2; ++rep) {
      int idx = t + (rep << 7);                // 0..255
      int n = idx >> 4, d = idx & 15;
      sB[(n << 4) + d] = (_Float16)x[(tile * 16 + n) * 16 + d];
    }
    float xr[8] = {0,0,0,0,0,0,0,0};
    float g [8] = {0,0,0,0,0,0,0,0};
    if (wv == 0 && lane < 16) {
      #pragma unroll
      for (int b = 0; b < 8; ++b) xr[b] = x[(tile * 16 + lane) * 16 + b];
    }
    __syncthreads();

    // ---------- phase 1: U1 = W1 * x ; h1 = tanh(U1+b1) -> sH1 panel ----------
    {
      v16h xb = *(const v16h*)(sB + (lane << 4));
      #pragma unroll
      for (int mi = 0; mi < 4; ++mi) {
        int m = (wv << 2) + mi;
        v16h a = {};
        const _Float16* pa = sW1c + (m << 8) + (n16 << 3);
        #pragma unroll
        for (int j = 0; j < 8; ++j) if (lane < 16) { a[j] = pa[j]; a[8 + j] = pa[128 + j]; }   // upper lane half (K 16..31) stays zero
        v8f acc = {};
        acc = wmma16(a, xb, acc);
        #pragma unroll
        for (int r = 0; r < 8; ++r) {
          int i = (m << 4) + r + hi8;
          float h1 = fast_tanh(acc[r] + sb1[i]);
          sH1[baddr(i, n16)] = (_Float16)h1;
        }
      }
    }
    __syncthreads();

    // ---------- phase 2: U2 = W2 * h1 ; P = w3s*s2 -> sB ; c2 -> sC2 ----------
    {
      v16h bf[8];
      #pragma unroll
      for (int f = 0; f < 8; ++f) bf[f] = *(const v16h*)(sH1 + (f << 9) + (lane << 4));
      #pragma unroll
      for (int mi = 0; mi < 4; ++mi) {
        int m = (wv << 2) + mi;
        v8f acc = {};
        #pragma unroll
        for (int f = 0; f < 8; ++f) {
          v16h a = *(const v16h*)(sW2A + (((m << 3) + f) << 9) + (lane << 4));
          acc = wmma16(a, bf[f], acc);
        }
        #pragma unroll
        for (int r = 0; r < 8; ++r) {
          int i = (m << 4) + r + hi8;
          float h2 = fast_tanh(acc[r] + sb2[i]);
          float s2 = 1.f - h2 * h2;
          float w3 = sw3[i];
          sB[baddr(i, n16)]   = (_Float16)(w3 * s2);
          sC2[(i << 4) + n16] = (_Float16)(-2.f * w3 * h2 * s2);
        }
      }
    }
    __syncthreads();

    // ---------- phase 3: Q = W2^T * P ; R = s1*Q -> sB ; cQ -> sCQ ----------
    {
      v16h bf[8];
      #pragma unroll
      for (int f = 0; f < 8; ++f) bf[f] = *(const v16h*)(sB + (f << 9) + (lane << 4));
      __syncthreads();                          // everyone hoisted P before sB is reused
      #pragma unroll
      for (int mi = 0; mi < 4; ++mi) {
        int m = (wv << 2) + mi;
        v8f acc = {};
        #pragma unroll
        for (int f = 0; f < 8; ++f) {
          v16h a = *(const v16h*)(sW2T + (((m << 3) + f) << 9) + (lane << 4));
          acc = wmma16(a, bf[f], acc);
        }
        #pragma unroll
        for (int r = 0; r < 8; ++r) {
          int i = (m << 4) + r + hi8;
          float q  = acc[r];
          float h1 = (float)sH1[baddr(i, n16)];
          float s1 = 1.f - h1 * h1;
          sB[baddr(i, n16)]   = (_Float16)(s1 * q);
          sCQ[(i << 4) + n16] = (_Float16)(-2.f * h1 * s1 * q);
        }
      }
    }
    __syncthreads();

    // ---------- phase 3b: g = W1^T * R  (wave 0) ----------
    if (wv == 0) {
      v8f acc = {};
      #pragma unroll
      for (int f = 0; f < 8; ++f) {
        v16h a = *(const v16h*)(sW1T + (f << 9) + (lane << 4));
        v16h b = *(const v16h*)(sB  + (f << 9) + (lane << 4));
        acc = wmma16(a, b, acc);
      }
      #pragma unroll
      for (int r = 0; r < 8; ++r) g[r] = acc[r];
    }

    // ---------- tangent loop: Hessian columns k = 8..15 ----------
    for (int kt = 0; kt < 8; ++kt) {
      int k = 8 + kt;
      __syncthreads();                          // sB free again (prev consumers done)
      // D1[i][n] = s1 * W1[i][k] ; each wave owns fragments 2wv, 2wv+1
      #pragma unroll
      for (int ff = 0; ff < 2; ++ff) {
        int f = (wv << 1) + ff;
        #pragma unroll
        for (int h = 0; h < 16; ++h) {
          int addr = (f << 9) + (lane << 4) + h;
          int i = (f << 5) + ((lane >> 4) << 4) + h;
          float h1 = (float)sH1[addr];
          float s1 = 1.f - h1 * h1;
          float w1k = (float)sW1c[((i >> 4) << 8) + ((i & 15) << 3) + 128 + (k & 7)];
          sB[addr] = (_Float16)(s1 * w1k);
        }
      }
      __syncthreads();

      // U2dot = W2 * D1 ; Pdot = c2 * U2dot -> sB
      {
        v16h bf[8];
        #pragma unroll
        for (int f = 0; f < 8; ++f) bf[f] = *(const v16h*)(sB + (f << 9) + (lane << 4));
        __syncthreads();
        #pragma unroll
        for (int mi = 0; mi < 4; ++mi) {
          int m = (wv << 2) + mi;
          v8f acc = {};
          #pragma unroll
          for (int f = 0; f < 8; ++f) {
            v16h a = *(const v16h*)(sW2A + (((m << 3) + f) << 9) + (lane << 4));
            acc = wmma16(a, bf[f], acc);
          }
          #pragma unroll
          for (int r = 0; r < 8; ++r) {
            int i = (m << 4) + r + hi8;
            sB[baddr(i, n16)] = (_Float16)((float)sC2[(i << 4) + n16] * acc[r]);
          }
        }
      }
      __syncthreads();

      // Qdot = W2^T * Pdot ; Rdot = cQ*W1[i][k] + s1*Qdot -> sB
      {
        v16h bf[8];
        #pragma unroll
        for (int f = 0; f < 8; ++f) bf[f] = *(const v16h*)(sB + (f << 9) + (lane << 4));
        __syncthreads();
        #pragma unroll
        for (int mi = 0; mi < 4; ++mi) {
          int m = (wv << 2) + mi;
          v8f acc = {};
          #pragma unroll
          for (int f = 0; f < 8; ++f) {
            v16h a = *(const v16h*)(sW2T + (((m << 3) + f) << 9) + (lane << 4));
            acc = wmma16(a, bf[f], acc);
          }
          #pragma unroll
          for (int r = 0; r < 8; ++r) {
            int i = (m << 4) + r + hi8;
            float h1 = (float)sH1[baddr(i, n16)];
            float s1 = 1.f - h1 * h1;
            float w1k = (float)sW1c[((i >> 4) << 8) + ((i & 15) << 3) + 128 + (k & 7)];
            sB[baddr(i, n16)] = (_Float16)((float)sCQ[(i << 4) + n16] * w1k + s1 * acc[r]);
          }
        }
      }
      __syncthreads();

      // H[:,k] = W1^T * Rdot  (wave 0); stash per-element columns in sHcol
      if (wv == 0) {
        v8f acc = {};
        #pragma unroll
        for (int f = 0; f < 8; ++f) {
          v16h a = *(const v16h*)(sW1T + (f << 9) + (lane << 4));
          v16h b = *(const v16h*)(sB  + (f << 9) + (lane << 4));
          acc = wmma16(a, b, acc);
        }
        int half = lane >> 4;                   // 0: rows 0..7 (dxdv), 1: rows 8..15 (dvdv)
        #pragma unroll
        for (int r = 0; r < 8; ++r)
          sHcol[((((n16 << 1) + half) << 3) + kt) * 8 + r] = acc[r];
      }
    }
    __syncthreads();

    // ---------- solve dvdvL * a = dxL - dxdvL * x[:8]  (wave 0, lanes 0..15) ----------
    float* S = (float*)sB;                      // 16 systems x (8 rows x 10 cols)
    if (wv == 0 && lane < 16) {
      int n = lane;
      const float* lo = sHcol + ((n << 1)    ) * 64;  // [kt][r] = H[r, 8+kt]   (r<8)
      const float* hp = sHcol + ((n << 1) + 1) * 64;  // [kt][r] = H[8+r, 8+kt]
      #pragma unroll
      for (int a_ = 0; a_ < 8; ++a_) {
        #pragma unroll
        for (int b_ = 0; b_ < 8; ++b_)
          S[n * 80 + a_ * 10 + b_] = hp[b_ * 8 + a_];  // dvdv[a][b]
        float rhs = g[a_];
        #pragma unroll
        for (int b_ = 0; b_ < 8; ++b_)
          rhs -= lo[a_ * 8 + b_] * xr[b_];             // dxdv[a][b] * x[b]
        S[n * 80 + a_ * 10 + 8] = rhs;
      }
      float* A = S + n * 80;                    // Gauss-Jordan, partial pivoting
      for (int p = 0; p < 8; ++p) {
        int piv = p; float best = fabsf(A[p * 10 + p]);
        for (int r = p + 1; r < 8; ++r) {
          float v = fabsf(A[r * 10 + p]);
          if (v > best) { best = v; piv = r; }
        }
        if (piv != p)
          for (int c = p; c < 9; ++c) {
            float tmp = A[p * 10 + c]; A[p * 10 + c] = A[piv * 10 + c]; A[piv * 10 + c] = tmp;
          }
        float inv = 1.f / A[p * 10 + p];
        for (int c = p; c < 9; ++c) A[p * 10 + c] *= inv;
        for (int r = 0; r < 8; ++r) {
          if (r == p) continue;
          float fac = A[r * 10 + p];
          if (fac != 0.f)
            for (int c = p; c < 9; ++c) A[r * 10 + c] -= fac * A[p * 10 + c];
        }
      }
      #pragma unroll
      for (int a_ = 0; a_ < 8; ++a_)
        out[(tile * 16 + n) * 8 + a_] = A[a_ * 10 + 8];
    }
    __syncthreads();
  }
}

extern "C" void kernel_launch(void* const* d_in, const int* in_sizes, int n_in,
                              void* d_out, int out_size, void* d_ws, size_t ws_size,
                              hipStream_t stream) {
  const float* x  = (const float*)d_in[0];
  const float* W1 = (const float*)d_in[1];
  const float* b1 = (const float*)d_in[2];
  const float* W2 = (const float*)d_in[3];
  const float* b2 = (const float*)d_in[4];
  const float* W3 = (const float*)d_in[5];
  (void)in_sizes; (void)n_in; (void)d_ws; (void)ws_size; (void)out_size;
  lnn_kernel<<<dim3(256), dim3(128), 0, stream>>>(x, W1, b1, W2, b2, W3, (float*)d_out);
}
